// ProvGNN_67577015435766
// MI455X (gfx1250) — compile-verified, matrix operand order checked by reading, not run on hardware
//
#include <hip/hip_runtime.h>
#include <hip/hip_bf16.h>

// ---------------------------------------------------------------------------
// GIN (3 layers, HID=128) on MI455X / gfx1250.
//  h = x * imp ; z = h
//  per layer: scatter_add(h[src] -> z[dst])          (HW f32 atomics, z in L2)
//             h = relu( relu(z@W1+b1)@W2 + b2 )*imp  (bf16 WMMA, fp32 accum)
//                 -> dual-stored to h and z (next layer), no copy kernel
//             pooled[l] = segment_sum(h, batch)      (batch sorted)
// A-tile staging uses global_load_async_to_lds_b128 (ASYNCcnt) overlapped with
// global_prefetch of the weight matrices.
// ---------------------------------------------------------------------------

#define N_NODES  50000
#define N_EDGES  600000
#define HID      128
#define N_LAYERS 3
#define N_GRAPHS 64
#define OUT_COLS (N_LAYERS * HID)   // 384

typedef __attribute__((ext_vector_type(16))) __bf16 v16bf;
typedef __attribute__((ext_vector_type(8)))  __bf16 v8bf;
typedef __attribute__((ext_vector_type(8)))  float  v8f;

// Guaranteed-hardware fp32 global atomic add (fire-and-forget, STOREcnt).
__device__ __forceinline__ void gadd_f32(float* p, float v) {
    asm volatile("global_atomic_add_f32 %0, %1, off" :: "v"(p), "v"(v) : "memory");
}

// Async global->LDS 16B copy (per-lane), tracked by ASYNCcnt.
__device__ __forceinline__ void async_g2l_b128(unsigned lds_addr, const void* gptr) {
    asm volatile("global_load_async_to_lds_b128 %0, %1, off"
                 :: "v"(lds_addr), "v"(gptr) : "memory");
}
__device__ __forceinline__ void wait_async0() {
    asm volatile("s_wait_asynccnt 0" ::: "memory");
}
// Low 32 bits of a flat pointer into the LDS aperture == LDS byte offset.
__device__ __forceinline__ unsigned lds_off(const void* p) {
    return (unsigned)(unsigned long long)p;
}

// Build a 16x32 bf16 WMMA fragment half-row: elements 0..7 are K..K+7,
// elements 8..15 are K+16..K+23 (per CDNA5 ISA 16-bit A/B VGPR layout).
__device__ __forceinline__ v16bf load_frag(const __bf16* base) {
    v8bf lo = *(const v8bf*)(base);
    v8bf hi = *(const v8bf*)(base + 16);
    v16bf r;
#pragma unroll
    for (int i = 0; i < 8; ++i) { r[i] = lo[i]; r[i + 8] = hi[i]; }
    return r;
}

// ---------------------------------------------------------------------------
// Prep: h = z = x * imp (float4), W1t/W2t = bf16 transpose of W1/W2, zero out.
// ---------------------------------------------------------------------------
__global__ __launch_bounds__(256) void gin_prep(
    const float* __restrict__ x, const float* __restrict__ imp,
    const float* __restrict__ W1, const float* __restrict__ W2,
    float* __restrict__ h, float* __restrict__ z,
    __bf16* __restrict__ W1t, __bf16* __restrict__ W2t,
    float* __restrict__ out)
{
    int i = blockIdx.x * 256 + threadIdx.x;          // float4 index
    if (i < N_NODES * HID / 4) {
        float4 v = ((const float4*)x)[i];
        float  s = imp[(i * 4) >> 7];
        float4 r; r.x = v.x * s; r.y = v.y * s; r.z = v.z * s; r.w = v.w * s;
        ((float4*)h)[i] = r;
        ((float4*)z)[i] = r;
    }
    if (i < N_LAYERS * HID * HID) {                  // transpose: Wt[l][n][k] = W[l][k][n]
        int layer = i >> 14, rem = i & 16383;
        int k = rem >> 7, n = rem & 127;
        int t = layer * HID * HID + n * HID + k;
        W1t[t] = (__bf16)W1[i];
        W2t[t] = (__bf16)W2[i];
    }
    if (i < N_GRAPHS * OUT_COLS) out[i] = 0.0f;
}

// ---------------------------------------------------------------------------
// Edge scatter: one wave per edge; lane reads float4 of h[src], atomically
// adds into z[dst]. 512B coalesced gather per edge; atomics resolve in L2.
// ---------------------------------------------------------------------------
__global__ __launch_bounds__(256) void gin_scatter(
    const float* __restrict__ h, float* __restrict__ z,
    const int* __restrict__ src, const int* __restrict__ dst)
{
    int t = blockIdx.x * 256 + threadIdx.x;
    int e = t >> 5;
    if (e >= N_EDGES) return;
    int part = (t & 31) * 4;
    int s = src[e], d = dst[e];
    float4 v = *(const float4*)(h + s * HID + part);
    float* p = z + d * HID + part;
    gadd_f32(p + 0, v.x); gadd_f32(p + 1, v.y);
    gadd_f32(p + 2, v.z); gadd_f32(p + 3, v.w);
}

// ---------------------------------------------------------------------------
// Fused 2-layer MLP via WMMA. Block = 16 nodes x 128 cols, 8 waves; each wave
// owns one 16-col N-tile for both GEMMs. K=128 -> 4 wmma per GEMM per wave.
// A tile async-staged into LDS; fp32 staging buffer aliased with Tbuf.
// ---------------------------------------------------------------------------
__global__ __launch_bounds__(256) void gin_mlp_wmma(
    const float*  __restrict__ z,   const __bf16* __restrict__ W1t,
    const float*  __restrict__ b1,  const __bf16* __restrict__ W2t,
    const float*  __restrict__ b2,  const float*  __restrict__ imp,
    float* __restrict__ hout, float* __restrict__ zout)
{
    __shared__ __align__(16) float  Afp[16 * HID];    // 8KB fp32 stage / Tbuf alias
    __shared__ __align__(16) __bf16 Abuf[16 * HID];   // 4KB bf16 A
    __shared__ float simp[16];
    __bf16* Tbuf = (__bf16*)Afp;                      // dead after convert pass

    const int tid = threadIdx.x;
    const int nodeBase = blockIdx.x * 16;
    const char* gtile = (const char*)(z + nodeBase * HID);

    // Kick off async LDS fill of the 8KB fp32 tile: 512 x b128, 2 per lane.
    {
        unsigned abase = lds_off(Afp);
        async_g2l_b128(abase + tid * 16,         gtile + tid * 16);
        async_g2l_b128(abase + (tid + 256) * 16, gtile + (tid + 256) * 16);
    }
    if (tid < 16) simp[tid] = imp[nodeBase + tid];
    // Overlap: prefetch this layer's weight rows while the copy is in flight.
    if (tid < 128) __builtin_prefetch(W1t + tid * HID, 0, 3);
    else           __builtin_prefetch(W2t + (tid - 128) * HID, 0, 3);
    wait_async0();
    __syncthreads();

    // LDS->LDS fp32 -> bf16 convert pass (8 elems/thread, b128 in / b128 out).
    {
        int base = tid * 8;
        float4 u = *(const float4*)&Afp[base];
        float4 v = *(const float4*)&Afp[base + 4];
        v8bf o;
        o[0]=(__bf16)u.x; o[1]=(__bf16)u.y; o[2]=(__bf16)u.z; o[3]=(__bf16)u.w;
        o[4]=(__bf16)v.x; o[5]=(__bf16)v.y; o[6]=(__bf16)v.z; o[7]=(__bf16)v.w;
        *(v8bf*)&Abuf[base] = o;
    }
    __syncthreads();

    const int wv   = tid >> 5;
    const int lane = tid & 31;
    const int half = lane >> 4;      // K half-select per 16-bit WMMA layout
    const int nsub = lane & 15;      // A row / B col within tile
    const int ncol = wv * 16 + nsub; // output column 0..127

    // GEMM1: z @ W1
    v8f c = {};
#pragma unroll
    for (int kt = 0; kt < 4; ++kt) {
        const int koff = kt * 32 + half * 8;
        v16bf a = load_frag(&Abuf[nsub * HID + koff]);
        v16bf b = load_frag(&W1t[ncol * HID + koff]);
        c = __builtin_amdgcn_wmma_f32_16x16x32_bf16(false, a, false, b,
                                                    (short)0, c, false, false);
    }
    const float bias1 = b1[ncol];
#pragma unroll
    for (int r = 0; r < 8; ++r) {            // C layout: row = r + 8*half, col = ncol
        float v = c[r] + bias1;
        v = v > 0.f ? v : 0.f;
        Tbuf[(r + half * 8) * HID + ncol] = (__bf16)v;
    }
    __syncthreads();

    // GEMM2: relu(...) @ W2
    v8f c2 = {};
#pragma unroll
    for (int kt = 0; kt < 4; ++kt) {
        const int koff = kt * 32 + half * 8;
        v16bf a = load_frag(&Tbuf[nsub * HID + koff]);
        v16bf b = load_frag(&W2t[ncol * HID + koff]);
        c2 = __builtin_amdgcn_wmma_f32_16x16x32_bf16(false, a, false, b,
                                                     (short)0, c2, false, false);
    }
    const float bias2 = b2[ncol];
#pragma unroll
    for (int r = 0; r < 8; ++r) {
        int   row = r + half * 8;
        float v = c2[r] + bias2;
        v = v > 0.f ? v : 0.f;
        v *= simp[row];                       // h = relu(z2) * imp
        int idx = (nodeBase + row) * HID + ncol;
        hout[idx] = v;                        // pooled / gathered next layer
        zout[idx] = v;                        // seed for next layer's scatter
    }
}

// ---------------------------------------------------------------------------
// Global add pool: batch is sorted, so run-length accumulate per column and
// flush via hardware atomic only on graph-id change / chunk end.
// ---------------------------------------------------------------------------
#define POOL_NODES 250
__global__ __launch_bounds__(128) void gin_pool(
    const float* __restrict__ h, const int* __restrict__ batch,
    float* __restrict__ out, int layer)
{
    const int col = threadIdx.x;                     // 0..127
    const int n0 = blockIdx.x * POOL_NODES;
    int n1 = n0 + POOL_NODES; if (n1 > N_NODES) n1 = N_NODES;
    float acc = 0.f;
    int curg = batch[n0];
    for (int n = n0; n < n1; ++n) {
        int g = batch[n];
        if (g != curg) {
            gadd_f32(&out[curg * OUT_COLS + layer * HID + col], acc);
            acc = 0.f; curg = g;
        }
        acc += h[n * HID + col];
    }
    gadd_f32(&out[curg * OUT_COLS + layer * HID + col], acc);
}

// ---------------------------------------------------------------------------
extern "C" void kernel_launch(void* const* d_in, const int* in_sizes, int n_in,
                              void* d_out, int out_size, void* d_ws, size_t ws_size,
                              hipStream_t stream)
{
    const float* x     = (const float*)d_in[0];
    const int*   eidx  = (const int*)  d_in[1];   // [2, N_EDGES]
    const int*   batch = (const int*)  d_in[2];
    const float* imp   = (const float*)d_in[3];
    const float* W1    = (const float*)d_in[4];
    const float* b1    = (const float*)d_in[5];
    const float* W2    = (const float*)d_in[6];
    const float* b2    = (const float*)d_in[7];
    float* out = (float*)d_out;

    char* ws = (char*)d_ws;
    const size_t HB = (size_t)N_NODES * HID * sizeof(float);   // 25.6 MB
    float*  h   = (float*)(ws);
    float*  z   = (float*)(ws + HB);
    __bf16* W1t = (__bf16*)(ws + 2 * HB);
    __bf16* W2t = (__bf16*)(ws + 2 * HB + (size_t)N_LAYERS * HID * HID * 2);

    const int elemBlocks    = (N_NODES * HID / 4 + 255) / 256;   // 6250
    const int scatterBlocks = (N_EDGES * 32 + 255) / 256;        // 75000
    const int mlpBlocks     = N_NODES / 16;                      // 3125
    const int poolBlocks    = (N_NODES + POOL_NODES - 1) / POOL_NODES; // 200

    gin_prep<<<elemBlocks, 256, 0, stream>>>(x, imp, W1, W2, h, z, W1t, W2t, out);

    for (int l = 0; l < N_LAYERS; ++l) {
        gin_scatter<<<scatterBlocks, 256, 0, stream>>>(h, z, eidx, eidx + N_EDGES);
        gin_mlp_wmma<<<mlpBlocks, 256, 0, stream>>>(
            z, W1t + l * HID * HID, b1 + l * HID,
            W2t + l * HID * HID, b2 + l * HID, imp, h, z);
        gin_pool<<<poolBlocks, 128, 0, stream>>>(h, batch, out, l);
    }
}
